// VectorQuantizer_27109833573046
// MI455X (gfx1250) — compile-verified, matrix operand order checked by reading, not run on hardware
//
#include <hip/hip_runtime.h>
#include <hip/hip_bf16.h>
#include <stdint.h>

// ---------------------------------------------------------------------------
// VQ codebook search on MI455X (gfx1250, wave32, WMMA)
//   score[n,k] = x_n . e_k - 0.5*||e_k||^2  (argmax == argmin of distance)
//   GEMM via v_wmma_f32_16x16x32_bf16, B staged to LDS via Tensor Data Mover,
//   B fragments software-pipelined so ds_load latency hides under WMMA.
// ---------------------------------------------------------------------------

#define N_ROWS      131072
#define D_DIM       256
#define K_CODES     1024
#define WG_THREADS  256
#define ROWS_PER_WG 128
#define CHUNK_COLS  64
#define N_CHUNKS    (K_CODES / CHUNK_COLS)     // 16
#define COL_STRIDE  528                        // 512B of bf16 data + 16B pad (bank spread)
#define BUF_BYTES   (CHUNK_COLS * COL_STRIDE)  // 33792
#define HN_OFF      (2 * BUF_BYTES)
#define WARR_OFF    (HN_OFF + K_CODES * 4)
#define SMEM_BYTES  (WARR_OFF + 64)            // 71744 bytes (dyn LDS; 320KB/WGP available)

typedef __attribute__((ext_vector_type(16))) __bf16 v16bf;
typedef __attribute__((ext_vector_type(8)))  float  v8f;

union Frag {
  v16bf    v;
  uint32_t u[8];
  uint4    q[2];
};

__device__ __forceinline__ uint32_t f2bf(float f) {
  uint32_t u = __float_as_uint(f);
  uint32_t r = u + 0x7FFFu + ((u >> 16) & 1u);   // RNE
  return r >> 16;
}

// ---------------------- Tensor Data Mover path (gfx1250) -------------------
#if defined(__gfx1250__) && __has_builtin(__builtin_amdgcn_tensor_load_to_lds)
#define TDM_PATH 1
typedef unsigned int tdm_u32x4 __attribute__((ext_vector_type(4)));
typedef int          tdm_i32x4 __attribute__((ext_vector_type(4)));
typedef int          tdm_i32x8 __attribute__((ext_vector_type(8)));

// Copy 64 codebook columns (64 x 512B contiguous in global) to LDS with
// 16B padding inserted every 512B (pad_interval=128dw, pad_amount=4dw).
__device__ __forceinline__ void tdm_load_chunk(const unsigned short* gsrc, void* ldst) {
  unsigned long long ga = (unsigned long long)(uintptr_t)gsrc;
  tdm_u32x4 g0;
  g0[0] = 1u;                                              // count=1 (valid user D#)
  g0[1] = (unsigned)(uintptr_t)ldst;                       // lds byte address (low 32 of flat)
  g0[2] = (unsigned)(ga & 0xFFFFFFFFull);                  // global_addr[31:0]
  g0[3] = (unsigned)((ga >> 32) & 0x1FFFFFFull) | (2u << 30);  // global_addr[56:32] | type=2
  tdm_i32x8 g1;
  g1[0] = (int)((3u << 16) | (1u << 20) | (6u << 22) | (3u << 25)); // data=8B, pad_en, int=128dw, amt=4dw
  g1[1] = (int)(64u << 16);                                // tensor_dim0 = 64 (8B units = 512B row)
  g1[2] = (int)(1024u << 16);                              // tensor_dim1 = 1024 (whole codebook)
  g1[3] = (int)(64u << 16);                                // tile_dim0 = 64
  g1[4] = (int)64;                                         // tile_dim1 = 64 cols, tile_dim2 = 0
  g1[5] = 64;                                              // tensor_dim0_stride = 64 (contiguous)
  g1[6] = 0;
  g1[7] = 0;
  tdm_i32x4 z4 = {0, 0, 0, 0};
  tdm_i32x8 z8 = {0, 0, 0, 0, 0, 0, 0, 0};
  // amdgpu-toolchain (clang-23 / therock-10.0) 6-arg form
  __builtin_amdgcn_tensor_load_to_lds(g0, g1, z4, z4, z8, 0);
}
#else
#define TDM_PATH 0
#endif

// ---------------------- prep: transpose codebook, norms --------------------
__global__ void __launch_bounds__(256)
vq_prep_kernel(const float* __restrict__ E, float* __restrict__ Etf,
               unsigned short* __restrict__ Ebf, float* __restrict__ hn,
               unsigned int* __restrict__ hist) {
  const int k = blockIdx.x;     // code index
  const int d = threadIdx.x;    // feature index
  float v = E[(size_t)d * K_CODES + k];
  Etf[(size_t)k * D_DIM + d] = v;
  Ebf[(size_t)k * D_DIM + d] = (unsigned short)f2bf(v);
  float s = v * v;
#pragma unroll
  for (int off = 1; off < 32; off <<= 1) s += __shfl_xor(s, off, 32);
  __shared__ float ps[8];
  if ((threadIdx.x & 31) == 0) ps[threadIdx.x >> 5] = s;
  __syncthreads();
  if (threadIdx.x == 0) {
    float t = 0.f;
#pragma unroll
    for (int i = 0; i < 8; ++i) t += ps[i];
    hn[k] = 0.5f * t;
    hist[k] = 0u;    // re-zeroed every launch (graph replay safe)
  }
}

// ---------------------- main: WMMA search + gather -------------------------
__global__ void __launch_bounds__(WG_THREADS)
vq_main_kernel(const float* __restrict__ X,
               const unsigned short* __restrict__ Ebf,
               const float* __restrict__ Etf,
               const float* __restrict__ hn_g,
               float* __restrict__ qout,
               float* __restrict__ indout,
               float* __restrict__ lossPart,
               unsigned int* __restrict__ hist) {
  extern __shared__ __align__(16) unsigned char smem[];
  unsigned char* buf0 = smem;
  unsigned char* buf1 = smem + BUF_BYTES;
  float* hns  = (float*)(smem + HN_OFF);
  float* warr = (float*)(smem + WARR_OFF);

  const int tid  = threadIdx.x;
  const int lane = tid & 31;
  const int wave = tid >> 5;
  const int l15  = lane & 15;
  const int hi   = lane >> 4;

  const int rowbase = blockIdx.x * ROWS_PER_WG + wave * 16;
  const int myrow   = rowbase + l15;

  for (int i = tid; i < K_CODES; i += WG_THREADS) hns[i] = hn_g[i];

  // A tile: 16 rows x 256 d as bf16 fragments, resident in 64 VGPRs/lane.
  // Layout per ISA 16-bit A 16x32: lane(l15)=row M; halves 0..7 = K d0..d0+7,
  // halves 8..15 = d0+16..d0+23; lanes 16..31 shifted by 8 in K.
  Frag A[8];
  {
    const float* xr = X + (size_t)myrow * D_DIM;
#pragma unroll
    for (int s = 0; s < 8; ++s) {
      const int d0 = 32 * s + hi * 8;
      float4 f0 = *(const float4*)(xr + d0);
      float4 f1 = *(const float4*)(xr + d0 + 4);
      float4 f2 = *(const float4*)(xr + d0 + 16);
      float4 f3 = *(const float4*)(xr + d0 + 20);
      A[s].u[0] = f2bf(f0.x) | (f2bf(f0.y) << 16);
      A[s].u[1] = f2bf(f0.z) | (f2bf(f0.w) << 16);
      A[s].u[2] = f2bf(f1.x) | (f2bf(f1.y) << 16);
      A[s].u[3] = f2bf(f1.z) | (f2bf(f1.w) << 16);
      A[s].u[4] = f2bf(f2.x) | (f2bf(f2.y) << 16);
      A[s].u[5] = f2bf(f2.z) | (f2bf(f2.w) << 16);
      A[s].u[6] = f2bf(f3.x) | (f2bf(f3.y) << 16);
      A[s].u[7] = f2bf(f3.z) | (f2bf(f3.w) << 16);
    }
  }

  float bestV[8];
  int   bestI[8];
#pragma unroll
  for (int v = 0; v < 8; ++v) { bestV[v] = -3.0e38f; bestI[v] = 0; }

#if TDM_PATH
  if (tid < 32) tdm_load_chunk(Ebf, buf0);   // prefetch chunk 0 (wave 0 issues)
#endif

  for (int c = 0; c < N_CHUNKS; ++c) {
    unsigned char* B = (c & 1) ? buf1 : buf0;
#if TDM_PATH
    if (c + 1 < N_CHUNKS) {
      __syncthreads();   // other buffer free (prev compute done); hns ready
      if (tid < 32) {
        tdm_load_chunk(Ebf + (size_t)(c + 1) * CHUNK_COLS * D_DIM,
                       (c & 1) ? buf0 : buf1);
        __builtin_amdgcn_s_wait_tensorcnt(1);  // chunk c landed, c+1 in flight
      }
    } else {
      if (tid < 32) __builtin_amdgcn_s_wait_tensorcnt(0);
    }
    __syncthreads();     // publish chunk c to all waves
#else
    __syncthreads();
    {
      const uint4* g = (const uint4*)(Ebf + (size_t)c * CHUNK_COLS * D_DIM);
#pragma unroll
      for (int i = 0; i < 8; ++i) {
        int li = tid + i * WG_THREADS;          // 2048 x 16B
        int col = li >> 5, u = li & 31;
        *(uint4*)(B + col * COL_STRIDE + u * 16) = g[li];
      }
    }
    __syncthreads();
#endif

    // 16x64 score tile, software-pipelined: fragments for d-step s+1 load
    // into a second register set while step s feeds the 4 WMMAs.
    v8f acc0 = {}, acc1 = {}, acc2 = {}, acc3 = {};
    const unsigned char* bbase = B + l15 * COL_STRIDE;
    Frag c0, c1, c2, c3, n0, n1, n2, n3;
    {
      const unsigned char* bp = bbase + hi * 16;   // s=0: doff = hi*8*2
      c0.q[0] = *(const uint4*)(bp);
      c0.q[1] = *(const uint4*)(bp + 32);
      c1.q[0] = *(const uint4*)(bp + 16 * COL_STRIDE);
      c1.q[1] = *(const uint4*)(bp + 16 * COL_STRIDE + 32);
      c2.q[0] = *(const uint4*)(bp + 32 * COL_STRIDE);
      c2.q[1] = *(const uint4*)(bp + 32 * COL_STRIDE + 32);
      c3.q[0] = *(const uint4*)(bp + 48 * COL_STRIDE);
      c3.q[1] = *(const uint4*)(bp + 48 * COL_STRIDE + 32);
    }
#pragma unroll
    for (int s = 0; s < 8; ++s) {
      if (s < 7) {
        const unsigned char* bp = bbase + (32 * (s + 1) + hi * 8) * 2;
        n0.q[0] = *(const uint4*)(bp);
        n0.q[1] = *(const uint4*)(bp + 32);
        n1.q[0] = *(const uint4*)(bp + 16 * COL_STRIDE);
        n1.q[1] = *(const uint4*)(bp + 16 * COL_STRIDE + 32);
        n2.q[0] = *(const uint4*)(bp + 32 * COL_STRIDE);
        n2.q[1] = *(const uint4*)(bp + 32 * COL_STRIDE + 32);
        n3.q[0] = *(const uint4*)(bp + 48 * COL_STRIDE);
        n3.q[1] = *(const uint4*)(bp + 48 * COL_STRIDE + 32);
      }
      acc0 = __builtin_amdgcn_wmma_f32_16x16x32_bf16(false, A[s].v, false, c0.v, (short)0, acc0, false, false);
      acc1 = __builtin_amdgcn_wmma_f32_16x16x32_bf16(false, A[s].v, false, c1.v, (short)0, acc1, false, false);
      acc2 = __builtin_amdgcn_wmma_f32_16x16x32_bf16(false, A[s].v, false, c2.v, (short)0, acc2, false, false);
      acc3 = __builtin_amdgcn_wmma_f32_16x16x32_bf16(false, A[s].v, false, c3.v, (short)0, acc3, false, false);
      if (s < 7) { c0 = n0; c1 = n1; c2 = n2; c3 = n3; }   // register rename after unroll
    }

    const int kc = c * CHUNK_COLS;
#pragma unroll
    for (int t = 0; t < 4; ++t) {
      const v8f a = (t == 0) ? acc0 : (t == 1) ? acc1 : (t == 2) ? acc2 : acc3;
      const int colabs = kc + t * 16 + l15;
      const float hn = hns[colabs];
#pragma unroll
      for (int v = 0; v < 8; ++v) {
        float sc = a[v] - hn;                 // = x.e - 0.5||e||^2
        if (sc > bestV[v]) { bestV[v] = sc; bestI[v] = colabs; }
      }
    }
  }

  // Cross-lane argmax over the 16 lanes holding the same row set.
#pragma unroll
  for (int off = 1; off < 16; off <<= 1) {
#pragma unroll
    for (int v = 0; v < 8; ++v) {
      float ov = __shfl_xor(bestV[v], off, 32);
      int   oi = __shfl_xor(bestI[v], off, 32);
      if (ov > bestV[v] || (ov == bestV[v] && oi < bestI[v])) { bestV[v] = ov; bestI[v] = oi; }
    }
  }

  // Gather f32 code rows -> quantize; exact f32 loss partials; indices; histogram.
  float lsum = 0.0f;
#pragma unroll
  for (int v = 0; v < 8; ++v) {
    const int r   = rowbase + hi * 8 + v;
    const int idx = bestI[v];
    if (l15 == v) {
      indout[r] = (float)idx;
      atomicAdd(&hist[idx], 1u);
    }
    const float* qv = Etf + (size_t)idx * D_DIM;
    const float* xv = X + (size_t)r * D_DIM;
    float*       ov = qout + (size_t)r * D_DIM;
    const int e0 = l15 * 16;
#pragma unroll
    for (int j = 0; j < 16; j += 4) {
      float4 q4 = *(const float4*)(qv + e0 + j);
      float4 x4 = *(const float4*)(xv + e0 + j);
      *(float4*)(ov + e0 + j) = q4;
      float dx = q4.x - x4.x, dy = q4.y - x4.y, dz = q4.z - x4.z, dw = q4.w - x4.w;
      lsum += dx * dx + dy * dy + dz * dz + dw * dw;
    }
  }
#pragma unroll
  for (int off = 1; off < 32; off <<= 1) lsum += __shfl_xor(lsum, off, 32);
  if (lane == 0) warr[wave] = lsum;
  __syncthreads();
  if (tid == 0) {
    float t = 0.f;
#pragma unroll
    for (int i = 0; i < 8; ++i) t += warr[i];
    lossPart[blockIdx.x] = t;   // deterministic: fixed slot per block
  }
}

// ---------------------- finalize: loss + perplexity ------------------------
__global__ void __launch_bounds__(1024)
vq_fin_kernel(const float* __restrict__ lossPart,
              const unsigned int* __restrict__ hist,
              float* __restrict__ out2) {
  __shared__ float psE[32], psL[32];
  const int t = threadIdx.x;                  // 1024 threads
  const float p = (float)hist[t] * (1.0f / (float)N_ROWS);
  float e = p * logf(p + 1e-10f);
  float l = lossPart[t];
#pragma unroll
  for (int off = 1; off < 32; off <<= 1) {
    e += __shfl_xor(e, off, 32);
    l += __shfl_xor(l, off, 32);
  }
  if ((t & 31) == 0) { psE[t >> 5] = e; psL[t >> 5] = l; }
  __syncthreads();
  if (t < 32) {
    float e2 = psE[t], l2 = psL[t];
#pragma unroll
    for (int off = 1; off < 32; off <<= 1) {
      e2 += __shfl_xor(e2, off, 32);
      l2 += __shfl_xor(l2, off, 32);
    }
    if (t == 0) {
      out2[0] = l2 * (1.0f / ((float)N_ROWS * (float)D_DIM));  // loss
      out2[1] = expf(-e2);                                      // perplexity
    }
  }
}

// ---------------------- launcher -------------------------------------------
extern "C" void kernel_launch(void* const* d_in, const int* in_sizes, int n_in,
                              void* d_out, int out_size, void* d_ws, size_t ws_size,
                              hipStream_t stream) {
  const float* X = (const float*)d_in[0];   // [131072, 256] f32
  const float* E = (const float*)d_in[1];   // [256, 1024] f32
  float* out = (float*)d_out;               // quantize | loss | perplexity | embed_ind

  char* ws = (char*)d_ws;
  float*          Etf      = (float*)ws;                                   // 1 MB  codebook^T f32
  unsigned short* Ebf      = (unsigned short*)(ws + (1u << 20));           // 512KB codebook^T bf16
  float*          hn       = (float*)(ws + (1u << 20) + (512u << 10));     // 4 KB  0.5*||e||^2
  float*          lossPart = hn + K_CODES;                                 // 4 KB  per-block loss
  unsigned int*   hist     = (unsigned int*)(lossPart + K_CODES);          // 4 KB  code histogram

  vq_prep_kernel<<<K_CODES, 256, 0, stream>>>(E, Etf, Ebf, hn, hist);

  const size_t NQ = (size_t)N_ROWS * D_DIM;
  vq_main_kernel<<<N_ROWS / ROWS_PER_WG, WG_THREADS, SMEM_BYTES, stream>>>(
      X, Ebf, Etf, hn, out, out + NQ + 2, lossPart, hist);

  vq_fin_kernel<<<1, K_CODES, 0, stream>>>(lossPart, hist, out + NQ);
}
